// AtlasMemoryPoly_43198781063790
// MI455X (gfx1250) — matrix operand order repaired and checked by reading, not compiled, not run on hardware
//
#include <hip/hip_runtime.h>
#include <hip/hip_bf16.h>
#include <stdint.h>

#define DIMX   256
#define HIDDEN 1024
#define TRIU   ((DIMX * (DIMX + 1)) / 2)   /* 32896 */
#define PD     (DIMX + TRIU)               /* 33152, divisible by 128 */
#define TOKENS 2048
#define W2N    (HIDDEN * PD)               /* 33,947,648 */
#define W1N    (DIMX * HIDDEN)             /* 262,144 */

typedef __attribute__((ext_vector_type(16))) __bf16 v16bf;
typedef __attribute__((ext_vector_type(8)))  float  v8f;
typedef unsigned short u16;
typedef unsigned int   u32;

// ---------------------------------------------------------------- utilities
__device__ __forceinline__ u16 f2bf(float f) {
    u32 u = __float_as_uint(f);
    u32 r = u + 0x7FFFu + ((u >> 16) & 1u);   // round-to-nearest-even
    return (u16)(r >> 16);
}

// Load A fragment for v_wmma_f32_16x16x32_bf16.
// Per-lane: 16 consecutive bf16 at p (VGPR0-7).
__device__ __forceinline__ v16bf loadA(const u16* p) {
    union { uint4 q[2]; v16bf v; } u;
    u.q[0] = *(const uint4*)(p);
    u.q[1] = *(const uint4*)(p + 8);
    return u.v;
}

// Load B fragment: per-lane 16 consecutive bf16 (K contiguous in memory).
__device__ __forceinline__ v16bf loadB(const u16* p) {
    union { uint4 q[2]; v16bf v; } u;
    u.q[0] = *(const uint4*)(p);
    u.q[1] = *(const uint4*)(p + 8);
    return u.v;
}

// ----------------------------------------------------- 1) weight conversion
__global__ __launch_bounds__(256) void convert_weights(
    const float* __restrict__ w2, const float* __restrict__ w3,
    const float* __restrict__ w1,
    u16* __restrict__ w2b, u16* __restrict__ w3b, u16* __restrict__ w1b)
{
    const long n4 = (2L * W2N + W1N) / 4;            // float4 groups
    long i4 = (long)blockIdx.x * blockDim.x + threadIdx.x;
    const long step = (long)gridDim.x * blockDim.x;
    for (; i4 < n4; i4 += step) {
        long i = i4 * 4;
        const float* src; u16* dst; long off;
        if (i < (long)W2N)            { src = w2; dst = w2b; off = i; }
        else if (i < 2L * W2N)        { src = w3; dst = w3b; off = i - W2N; }
        else                          { src = w1; dst = w1b; off = i - 2L * W2N; }
        float4 f = *(const float4*)(src + off);
        u32 lo = (u32)f2bf(f.x) | ((u32)f2bf(f.y) << 16);
        u32 hi = (u32)f2bf(f.z) | ((u32)f2bf(f.w) << 16);
        uint2 pk; pk.x = lo; pk.y = hi;
        *(uint2*)(dst + off) = pk;
    }
}

// ------------------------------------------------- 2) poly feature build
__device__ __forceinline__ float poly_feat(const float* xs, int p) {
    if (p < DIMX) return xs[p];
    int t = p - DIMX;
    int i = (int)floorf((513.0f - sqrtf(263169.0f - 8.0f * (float)t)) * 0.5f);
    if (i < 0) i = 0;
    if (i > DIMX - 1) i = DIMX - 1;
    while (i > 0 && t < (i * (513 - i)) / 2) --i;
    while (i < DIMX - 1 && t >= ((i + 1) * (513 - (i + 1))) / 2) ++i;
    int off = (i * (513 - i)) / 2;
    int j = i + (t - off);
    return xs[i] * xs[j];
}

__global__ __launch_bounds__(256) void build_poly(
    const float* __restrict__ x, u16* __restrict__ xp)
{
    __shared__ float xs[DIMX];
    const int t = blockIdx.y;

#if __has_builtin(__builtin_amdgcn_global_load_async_to_lds_b32) && \
    __has_builtin(__builtin_amdgcn_s_wait_asynccnt)
    {
        // DMA the x row straight into LDS (no VGPR round-trip), one b32/lane.
        // Builtin signature on this toolchain uses generic int* params.
        const float* gp = x + (long)t * DIMX + threadIdx.x;
        float* lp = &xs[threadIdx.x];
        __builtin_amdgcn_global_load_async_to_lds_b32((int*)gp, (int*)lp, 0, 0);
        __builtin_amdgcn_s_wait_asynccnt(0);
    }
#else
    xs[threadIdx.x] = x[(long)t * DIMX + threadIdx.x];   // blockDim == 256 == DIMX
#endif
    __syncthreads();

    long p0 = (long)blockIdx.x * (256 * 8) + (long)threadIdx.x * 8;
    u16* orow = xp + (long)t * PD;
    #pragma unroll
    for (int e = 0; e < 8; e += 2) {
        long p = p0 + e;
        if (p >= PD) break;
        float f0 = poly_feat(xs, (int)p);
        float f1 = poly_feat(xs, (int)p + 1);            // PD even: p+1 valid
        u32 pk = (u32)f2bf(f0) | ((u32)f2bf(f1) << 16);
        *(u32*)(orow + p) = pk;                          // p even -> 4B aligned
    }
}

// -------------------------------------- 3) main WMMA GEMM: gate & value
// Tile: M_TILE=128 x N_TILE=128 per 16-wave (512-thread) workgroup.
// Wave (wq,wr) owns a 2m x 2n quad of 16x16 tiles for both matmuls:
//   8 accumulators (64 VGPRs), 6 live fragments, 12 b128 loads : 8 WMMAs.
__global__ __launch_bounds__(512) void gemm_gate_value(
    const u16* __restrict__ xp, const u16* __restrict__ w2b,
    const u16* __restrict__ w3b, u16* __restrict__ gated)
{
    const int wave = threadIdx.x >> 5;          // 0..15
    const int lane = threadIdx.x & 31;
    const int col  = lane & 15;
    const int hi   = lane >> 4;                 // half-wave select
    const int wq   = wave >> 2;                 // 0..3 -> m-pair
    const int wr   = wave & 3;                  // 0..3 -> n-pair
    const int mb   = blockIdx.x * 128 + wq * 32;
    const int n0   = blockIdx.y * 128 + wr * 32;

    const u16* arow[2];
    const u16* b2row[2];
    const u16* b3row[2];
    #pragma unroll
    for (int i = 0; i < 2; ++i) {
        arow[i]  = xp  + (long)(mb + i * 16 + col) * PD;
        b2row[i] = w2b + (long)(n0 + i * 16 + col) * PD;
        b3row[i] = w3b + (long)(n0 + i * 16 + col) * PD;
    }

    v8f accG[2][2] = {{{0}}};
    v8f accV[2][2] = {{{0}}};

    #pragma unroll 2
    for (int kb = 0; kb < PD; kb += 32) {
        const int ka  = kb + hi * 16;           // A per-lane K start
        const int kbb = kb + hi * 16;           // B per-lane K start
        __builtin_prefetch(b2row[0] + kbb + 128, 0, 1);
        __builtin_prefetch(b3row[0] + kbb + 128, 0, 1);
        v16bf A0 = loadA(arow[0] + ka);
        v16bf A1 = loadA(arow[1] + ka);
        v16bf B2a = loadB(b2row[0] + kbb);
        v16bf B2b = loadB(b2row[1] + kbb);
        v16bf B3a = loadB(b3row[0] + kbb);
        v16bf B3b = loadB(b3row[1] + kbb);
        accG[0][0] = __builtin_amdgcn_wmma_f32_16x16x32_bf16(false, A0, false, B2a, (short)0, accG[0][0], false, false);
        accG[0][1] = __builtin_amdgcn_wmma_f32_16x16x32_bf16(false, A0, false, B2b, (short)0, accG[0][1], false, false);
        accV[0][0] = __builtin_amdgcn_wmma_f32_16x16x32_bf16(false, A0, false, B3a, (short)0, accV[0][0], false, false);
        accV[0][1] = __builtin_amdgcn_wmma_f32_16x16x32_bf16(false, A0, false, B3b, (short)0, accV[0][1], false, false);
        accG[1][0] = __builtin_amdgcn_wmma_f32_16x16x32_bf16(false, A1, false, B2a, (short)0, accG[1][0], false, false);
        accG[1][1] = __builtin_amdgcn_wmma_f32_16x16x32_bf16(false, A1, false, B2b, (short)0, accG[1][1], false, false);
        accV[1][0] = __builtin_amdgcn_wmma_f32_16x16x32_bf16(false, A1, false, B3a, (short)0, accV[1][0], false, false);
        accV[1][1] = __builtin_amdgcn_wmma_f32_16x16x32_bf16(false, A1, false, B3b, (short)0, accV[1][1], false, false);
    }

    // epilogue: gated = silu(gate) * value, bf16
    #pragma unroll
    for (int im = 0; im < 2; ++im) {
        #pragma unroll
        for (int in = 0; in < 2; ++in) {
            const int h = n0 + in * 16 + col;
            #pragma unroll
            for (int e = 0; e < 8; ++e) {
                const int tok = mb + im * 16 + e + hi * 8;  // C layout: M = e (+8 for hi)
                float g = accG[im][in][e];
                float v = accV[im][in][e];
                float s = g / (1.0f + __expf(-g));          // silu
                gated[(long)tok * HIDDEN + h] = f2bf(s * v);
            }
        }
    }
}

// -------------------------------- 4) output GEMM: out = x + gated @ w1^T
__global__ __launch_bounds__(256) void gemm_out(
    const u16* __restrict__ gated, const u16* __restrict__ w1b,
    const float* __restrict__ x, float* __restrict__ out)
{
    const int wave = threadIdx.x >> 5;
    const int lane = threadIdx.x & 31;
    const int col  = lane & 15;
    const int hi   = lane >> 4;
    const int mb   = blockIdx.x * 32;
    const int nA   = wave * 32;
    const int nB   = nA + 16;

    const u16* arow[2] = { gated + (long)(mb + col) * HIDDEN,
                           gated + (long)(mb + 16 + col) * HIDDEN };
    const u16* brow[2] = { w1b + (long)(nA + col) * HIDDEN,
                           w1b + (long)(nB + col) * HIDDEN };

    v8f acc[2][2] = {{{0}}};

    #pragma unroll 2
    for (int kb = 0; kb < HIDDEN; kb += 32) {
        v16bf B0 = loadB(brow[0] + kb + hi * 16);
        v16bf B1 = loadB(brow[1] + kb + hi * 16);
        #pragma unroll
        for (int m = 0; m < 2; ++m) {
            v16bf A = loadA(arow[m] + kb + hi * 16);
            acc[m][0] = __builtin_amdgcn_wmma_f32_16x16x32_bf16(
                false, A, false, B0, (short)0, acc[m][0], false, false);
            acc[m][1] = __builtin_amdgcn_wmma_f32_16x16x32_bf16(
                false, A, false, B1, (short)0, acc[m][1], false, false);
        }
    }

    #pragma unroll
    for (int m = 0; m < 2; ++m) {
        #pragma unroll
        for (int n = 0; n < 2; ++n) {
            const int d = (n ? nB : nA) + col;
            #pragma unroll
            for (int e = 0; e < 8; ++e) {
                const int tok = mb + m * 16 + e + hi * 8;
                out[(long)tok * DIMX + d] = x[(long)tok * DIMX + d] + acc[m][n][e];
            }
        }
    }
}

// --------------------------------------------------------------- launcher
extern "C" void kernel_launch(void* const* d_in, const int* in_sizes, int n_in,
                              void* d_out, int out_size, void* d_ws, size_t ws_size,
                              hipStream_t stream) {
    const float* x  = (const float*)d_in[0];
    const float* w1 = (const float*)d_in[1];
    const float* w2 = (const float*)d_in[2];
    const float* w3 = (const float*)d_in[3];

    u16* w2b   = (u16*)d_ws;
    u16* w3b   = w2b + (size_t)W2N;
    u16* w1b   = w3b + (size_t)W2N;
    u16* xp    = w1b + (size_t)W1N;
    u16* gated = xp + (size_t)TOKENS * PD;
    float* out = (float*)d_out;

    convert_weights<<<4096, 256, 0, stream>>>(w2, w3, w1, w2b, w3b, w1b);
    build_poly<<<dim3((PD + 2047) / 2048, TOKENS), 256, 0, stream>>>(x, xp);
    gemm_gate_value<<<dim3(TOKENS / 128, HIDDEN / 128), 512, 0, stream>>>(xp, w2b, w3b, gated);
    gemm_out<<<TOKENS / 32, 256, 0, stream>>>(gated, w1b, x, out);
}
